// NanoGPT_69518340653549
// MI455X (gfx1250) — matrix operand order repaired and checked by reading, not compiled, not run on hardware
//
#include <hip/hip_runtime.h>
#include <hip/hip_bf16.h>
#include <math.h>

// ---------------------------------------------------------------------------
// NanoGPT forward for MI455X (gfx1250): wave32, WMMA f32_16x16x32_f16,
// async global->LDS DMA (ASYNCcnt) for f16 tile staging.
// Residual stream x stays fp32; all WMMA operands staged/stored as f16.
// ---------------------------------------------------------------------------

typedef __attribute__((ext_vector_type(16))) _Float16 v16h;
typedef __attribute__((ext_vector_type(8)))  float    v8f;

#define D_MODEL 1024
#define T_SEQ   1024
#define N_HEAD  16
#define HEAD_SZ 64
#define N_LAYER 12
#define N_VOCAB 50257
#define N_ROWS  2048        // B*T
#define D_FF    4096
#define LN_EPS  1e-5f
#define ATT_SCALE 0.03125f  // D_MODEL^-0.5 (reference scales by n_embd^-0.5)

// ---- CDNA5 async global->LDS copy (tracked by ASYNCcnt) --------------------
// Copies 16 bytes per lane: LDS[dst32] = MEM[src64]. LDS aperture flat
// pointers carry the LDS offset in their low 32 bits, so truncation of a
// generic pointer to __shared__ yields a valid VDST address.
__device__ __forceinline__ void async_ld_lds_b128(unsigned lds_addr,
                                                  const void* gaddr) {
  asm volatile("global_load_async_to_lds_b128 %0, %1, off"
               :: "v"(lds_addr), "v"(gaddr) : "memory");
}
__device__ __forceinline__ void wait_async0() {
  asm volatile("s_wait_asynccnt 0x0" ::: "memory");
}
__device__ __forceinline__ unsigned lds_addr_of(const void* p) {
  return (unsigned)(uintptr_t)p;
}

// ---- WMMA fragment loaders (CDNA5 ISA 7.12.2 layouts) ----------------------
// A-matrix 16x32 f16: lane m = lane&15, hi = lane>>4.
// v16h element 8g+i  <->  K = g*16 + hi*8 + i   (g in {0,1}, i in 0..7)
__device__ __forceinline__ v16h ld_frag_a(const _Float16* p, int hi) {
  v16h a;
  const _Float16* p0 = p + hi * 8;
  const _Float16* p1 = p + 16 + hi * 8;
#pragma unroll
  for (int i = 0; i < 8; ++i) { a[i] = p0[i]; a[8 + i] = p1[i]; }
  return a;
}
// B-matrix 32x16 f16, stored contraction-contiguous per column in LDS:
// v16h element 8g+i  <->  K = g*16 + hi*8 + i   (g in {0,1}, i in 0..7)
__device__ __forceinline__ v16h ld_frag_b(const _Float16* p, int hi) {
  v16h b;
  const _Float16* q = p + hi * 8;
#pragma unroll
  for (int i = 0; i < 8; ++i) { b[i] = q[i]; b[8 + i] = q[16 + i]; }
  return b;
}

__device__ __forceinline__ v8f wmma_f16(v16h a, v16h b, v8f c) {
  return __builtin_amdgcn_wmma_f32_16x16x32_f16(false, a, false, b,
                                                (short)0, c, false, false);
}

// ---------------------------------------------------------------------------
// Generic GEMM: C[M,N] = epilogue(A[M,K](f16) x B[K,N](fp32)) (+bias,GELU,+res)
// Block: 256 threads = 8 waves; tile 128x128, BK = 32 (one WMMA K-step).
// Double-buffered LDS; A tiles staged by async DMA, B tiles converted+transposed.
// Requires M % 128 == 0, K % 32 == 0 (true for all launches here).
// ---------------------------------------------------------------------------
template <bool BIAS, bool RESID, bool GELU, typename OutT>
__global__ __launch_bounds__(256)
void gemm_wmma(const _Float16* __restrict__ A, const float* __restrict__ B,
               const float* __restrict__ bias, const float* __restrict__ resid,
               OutT* __restrict__ C, int M, int N, int K) {
  constexpr int BM = 128, BN = 128, BK = 32;
  constexpr int SA = BK + 8;   // half stride; rows stay 16B aligned
  constexpr int SB = BK + 8;
  __shared__ _Float16 Alds[2][BM * SA];
  __shared__ _Float16 Blds[2][BN * SB];   // transposed: [n][k]

  const int tid  = threadIdx.x;
  const int lane = tid & 31;
  const int wave = tid >> 5;
  const int wm   = wave & 3;   // 0..3 -> 32-row slab
  const int wn   = wave >> 2;  // 0..1 -> 64-col slab
  const int bm   = blockIdx.x * BM;
  const int bn   = blockIdx.y * BN;
  const int hi   = lane >> 4;
  const int l16  = lane & 15;
  const bool n_vec4 = ((N & 3) == 0);

  // Async DMA of A tile: 128x32 halves = 512 x 16B chunks, 2 per thread.
  auto stage_A = [&](int k0, int buf) {
#pragma unroll
    for (int i = 0; i < 2; ++i) {
      const int c   = tid + 256 * i;
      const int row = c >> 2;           // 0..127
      const int off = (c & 3) * 8;      // 0,8,16,24 halves
      const unsigned dst = lds_addr_of(&Alds[buf][row * SA + off]);
      const void* src = (const void*)(A + (size_t)(bm + row) * K + k0 + off);
      async_ld_lds_b128(dst, src);
    }
  };
  // B tile 32x128 fp32 -> f16, transposed into [n][k].
  auto stage_B = [&](int k0, int buf) {
    const int kr = tid >> 5;         // 0..7
    const int c4 = (tid & 31) * 4;   // 0..124
#pragma unroll
    for (int rr = 0; rr < 4; ++rr) {
      const int krow = kr + rr * 8;
      const int gcol = bn + c4;
      float4 f;
      if (n_vec4 && gcol + 3 < N) {
        f = *(const float4*)(B + (size_t)(k0 + krow) * N + gcol);
      } else {
        float t0 = (gcol + 0 < N) ? B[(size_t)(k0 + krow) * N + gcol + 0] : 0.f;
        float t1 = (gcol + 1 < N) ? B[(size_t)(k0 + krow) * N + gcol + 1] : 0.f;
        float t2 = (gcol + 2 < N) ? B[(size_t)(k0 + krow) * N + gcol + 2] : 0.f;
        float t3 = (gcol + 3 < N) ? B[(size_t)(k0 + krow) * N + gcol + 3] : 0.f;
        f = make_float4(t0, t1, t2, t3);
      }
      _Float16* bl = &Blds[buf][0];
      bl[(c4 + 0) * SB + krow] = (_Float16)f.x;
      bl[(c4 + 1) * SB + krow] = (_Float16)f.y;
      bl[(c4 + 2) * SB + krow] = (_Float16)f.z;
      bl[(c4 + 3) * SB + krow] = (_Float16)f.w;
    }
  };

  v8f acc[2][4];
  const v8f vzero = {0.f, 0.f, 0.f, 0.f, 0.f, 0.f, 0.f, 0.f};
#pragma unroll
  for (int i = 0; i < 2; ++i)
#pragma unroll
    for (int j = 0; j < 4; ++j) acc[i][j] = vzero;

  stage_A(0, 0);
  stage_B(0, 0);
  wait_async0();
  __syncthreads();

  const int nk = K / BK;
  for (int kt = 0; kt < nk; ++kt) {
    const int cur = kt & 1;
    if (kt + 1 < nk) {
      stage_A((kt + 1) * BK, cur ^ 1);
      stage_B((kt + 1) * BK, cur ^ 1);
      if (kt + 2 < nk)   // hint L2 for the B tile after next
        __builtin_prefetch((const void*)(B + (size_t)(kt + 2) * BK * N + bn + lane * 4), 0, 0);
    }

    v16h afr[2], bfr[4];
#pragma unroll
    for (int i = 0; i < 2; ++i)
      afr[i] = ld_frag_a(&Alds[cur][(wm * 32 + i * 16 + l16) * SA], hi);
#pragma unroll
    for (int j = 0; j < 4; ++j)
      bfr[j] = ld_frag_b(&Blds[cur][(wn * 64 + j * 16 + l16) * SB], hi);
#pragma unroll
    for (int i = 0; i < 2; ++i)
#pragma unroll
      for (int j = 0; j < 4; ++j)
        acc[i][j] = wmma_f16(afr[i], bfr[j], acc[i][j]);

    wait_async0();      // next tile's async DMA done (barrier won't wait it)
    __syncthreads();    // all waves done reading `cur`, B stores visible
  }

  // ---- epilogue: C/D layout row = v + 8*hi, col = lane&15 ----
#pragma unroll
  for (int i = 0; i < 2; ++i) {
    const int rowb = bm + wm * 32 + i * 16 + hi * 8;
#pragma unroll
    for (int j = 0; j < 4; ++j) {
      const int col = bn + wn * 64 + j * 16 + l16;
      if (col < N) {
#pragma unroll
        for (int v = 0; v < 8; ++v) {
          const int row = rowb + v;
          float val = acc[i][j][v];
          if (BIAS) val += bias[col];
          if (GELU) val = 0.5f * val * (1.0f + erff(val * 0.70710678118f));
          if (RESID) val += resid[(size_t)row * N + col];
          C[(size_t)row * N + col] = (OutT)val;
        }
      }
    }
  }
}

// ---------------------------------------------------------------------------
// Flash attention (causal). Grid: (T/128, B*H). Block: 256 thr = 8 waves.
// Q/K/V/att are f16. K tiles async-DMA'd to LDS; V transposed via VGPRs.
// ---------------------------------------------------------------------------
__global__ __launch_bounds__(256)
void flash_attn(const _Float16* __restrict__ Q, const _Float16* __restrict__ Kg,
                const _Float16* __restrict__ Vg, _Float16* __restrict__ O) {
  constexpr int BKV = 64;
  constexpr int SK = HEAD_SZ + 8;    // Klds[key][d]
  constexpr int SV = BKV + 8;        // Vtlds[d][key]
  constexpr int SP = BKV + 8;        // Plds[qrow][key]
  __shared__ _Float16 Klds[BKV * SK];
  __shared__ _Float16 Vtlds[HEAD_SZ * SV];
  __shared__ _Float16 Plds[128 * SP];

  const int qt   = blockIdx.x;            // query tile (128 rows)
  const int bh   = blockIdx.y;
  const int b    = bh >> 4;
  const int h    = bh & 15;
  const int tid  = threadIdx.x;
  const int lane = tid & 31;
  const int wave = tid >> 5;
  const int hi   = lane >> 4;
  const int l16  = lane & 15;

  const _Float16* Qp = Q + (size_t)b * T_SEQ * D_MODEL + h * HEAD_SZ;
  const _Float16* Kp = Kg + (size_t)b * T_SEQ * D_MODEL + h * HEAD_SZ;
  const _Float16* Vp = Vg + (size_t)b * T_SEQ * D_MODEL + h * HEAD_SZ;
  _Float16*       Op = O + (size_t)b * T_SEQ * D_MODEL + h * HEAD_SZ;

  // Q fragments (A-layout) for this wave's 16 rows; contraction HS=64.
  const int mrow = qt * 128 + wave * 16 + l16;
  const _Float16* qrow = Qp + (size_t)mrow * D_MODEL;
  v16h qf[2];
#pragma unroll
  for (int ks = 0; ks < 2; ++ks) {
#pragma unroll
    for (int i = 0; i < 8; ++i) {
      qf[ks][i]     = qrow[ks * 32 + hi * 8 + i];
      qf[ks][8 + i] = qrow[ks * 32 + 16 + hi * 8 + i];
    }
  }

  float m_i[8], l_i[8];
  v8f o_acc[4];
  const v8f vzero = {0.f, 0.f, 0.f, 0.f, 0.f, 0.f, 0.f, 0.f};
#pragma unroll
  for (int v = 0; v < 8; ++v) { m_i[v] = -3.0e38f; l_i[v] = 0.f; }
#pragma unroll
  for (int t = 0; t < 4; ++t) o_acc[t] = vzero;

  const int jmax = 2 * qt + 1;   // causal: keys up to (qt*128+127)
  for (int j = 0; j <= jmax; ++j) {
    // ---- K tile: async DMA, 64x64 halves = 512 x 16B chunks ----
#pragma unroll
    for (int i = 0; i < 2; ++i) {
      const int c   = tid + 256 * i;
      const int key = c >> 3;        // 0..63
      const int off = (c & 7) * 8;   // 0..56 halves
      const unsigned dst = lds_addr_of(&Klds[key * SK + off]);
      const void* src = (const void*)(Kp + (size_t)(j * BKV + key) * D_MODEL + off);
      async_ld_lds_b128(dst, src);
    }
    // ---- V tile: load f16 rows, scatter transposed [d][key] ----
#pragma unroll
    for (int i = 0; i < 2; ++i) {
      const int c    = tid + 256 * i;
      const int key  = c >> 3;
      const int doff = (c & 7) * 8;
      const _Float16* src = Vp + (size_t)(j * BKV + key) * D_MODEL + doff;
      _Float16 tmp[8];
#pragma unroll
      for (int ii = 0; ii < 8; ++ii) tmp[ii] = src[ii];
#pragma unroll
      for (int ii = 0; ii < 8; ++ii) Vtlds[(doff + ii) * SV + key] = tmp[ii];
    }
    wait_async0();
    __syncthreads();

    // ---- S = Q K^T : 4 key tiles x 2 K-steps ----
    v8f s[4];
#pragma unroll
    for (int jn = 0; jn < 4; ++jn) s[jn] = vzero;
#pragma unroll
    for (int ks = 0; ks < 2; ++ks) {
#pragma unroll
      for (int jn = 0; jn < 4; ++jn) {
        v16h bfr = ld_frag_b(&Klds[(jn * 16 + l16) * SK + ks * 32], hi);
        s[jn] = wmma_f16(qf[ks], bfr, s[jn]);
      }
    }

    // ---- scale + causal mask (branchless) ----
#pragma unroll
    for (int jn = 0; jn < 4; ++jn) {
      const int kglob = j * BKV + jn * 16 + l16;
#pragma unroll
      for (int v = 0; v < 8; ++v) {
        const int qglob = qt * 128 + wave * 16 + v + 8 * hi;
        float sv = s[jn][v] * ATT_SCALE;
        if (kglob > qglob) sv = -3.0e38f;
        s[jn][v] = sv;
      }
    }

    // ---- online softmax: row groups live in 16-lane halves ----
#pragma unroll
    for (int v = 0; v < 8; ++v) {
      float mx = -3.0e38f;
#pragma unroll
      for (int jn = 0; jn < 4; ++jn) mx = fmaxf(mx, s[jn][v]);
#pragma unroll
      for (int off = 1; off < 16; off <<= 1) mx = fmaxf(mx, __shfl_xor(mx, off, 32));
      const float mn = fmaxf(m_i[v], mx);
      const float corr = __expf(m_i[v] - mn);
      m_i[v] = mn;
      float rs = 0.f;
#pragma unroll
      for (int jn = 0; jn < 4; ++jn) {
        const float p = __expf(s[jn][v] - mn);
        s[jn][v] = p;
        rs += p;
      }
#pragma unroll
      for (int off = 1; off < 16; off <<= 1) rs += __shfl_xor(rs, off, 32);
      l_i[v] = l_i[v] * corr + rs;
#pragma unroll
      for (int t = 0; t < 4; ++t) o_acc[t][v] *= corr;
    }

    // ---- spill P (C-layout) to LDS so it can be re-read in A-layout ----
#pragma unroll
    for (int jn = 0; jn < 4; ++jn)
#pragma unroll
      for (int v = 0; v < 8; ++v)
        Plds[(wave * 16 + v + 8 * hi) * SP + jn * 16 + l16] = (_Float16)s[jn][v];
    __syncthreads();

    // ---- O += P x V : contraction over 64 keys = 2 K-steps ----
#pragma unroll
    for (int ks = 0; ks < 2; ++ks) {
      v16h pf = ld_frag_a(&Plds[(wave * 16 + l16) * SP + ks * 32], hi);
#pragma unroll
      for (int t = 0; t < 4; ++t) {
        v16h vf = ld_frag_b(&Vtlds[(t * 16 + l16) * SV + ks * 32], hi);
        o_acc[t] = wmma_f16(pf, vf, o_acc[t]);
      }
    }
    __syncthreads();   // before restaging K/V next iteration
  }

  // ---- normalize and store (f16) ----
#pragma unroll
  for (int t = 0; t < 4; ++t) {
#pragma unroll
    for (int v = 0; v < 8; ++v) {
      const int r = qt * 128 + wave * 16 + v + 8 * hi;
      const int c = t * 16 + l16;
      Op[(size_t)r * D_MODEL + c] = (_Float16)(o_acc[t][v] / l_i[v]);
    }
  }
}

// ---------------------------------------------------------------------------
// LayerNorm: one 256-thread block per row of 1024. fp32 in, f16 out.
// ---------------------------------------------------------------------------
__global__ __launch_bounds__(256)
void layernorm(const float* __restrict__ X, const float* __restrict__ g,
               const float* __restrict__ bta, _Float16* __restrict__ Y) {
  __shared__ float red[8];
  const int row = blockIdx.x;
  const int tid = threadIdx.x;
  const float* x = X + (size_t)row * D_MODEL;
  float xv[4];
#pragma unroll
  for (int i = 0; i < 4; ++i) xv[i] = x[tid + i * 256];

  float s = xv[0] + xv[1] + xv[2] + xv[3];
#pragma unroll
  for (int off = 16; off > 0; off >>= 1) s += __shfl_xor(s, off, 32);
  if ((tid & 31) == 0) red[tid >> 5] = s;
  __syncthreads();
  float tot = 0.f;
#pragma unroll
  for (int i = 0; i < 8; ++i) tot += red[i];
  const float mean = tot * (1.0f / D_MODEL);
  __syncthreads();

  float vs = 0.f;
#pragma unroll
  for (int i = 0; i < 4; ++i) { const float d = xv[i] - mean; vs += d * d; }
#pragma unroll
  for (int off = 16; off > 0; off >>= 1) vs += __shfl_xor(vs, off, 32);
  if ((tid & 31) == 0) red[tid >> 5] = vs;
  __syncthreads();
  float vtot = 0.f;
#pragma unroll
  for (int i = 0; i < 8; ++i) vtot += red[i];
  const float rstd = rsqrtf(vtot * (1.0f / D_MODEL) + LN_EPS);

  _Float16* y = Y + (size_t)row * D_MODEL;
#pragma unroll
  for (int i = 0; i < 4; ++i) {
    const int c = tid + i * 256;
    y[c] = (_Float16)((xv[i] - mean) * rstd * g[c] + bta[c]);
  }
}

// ---------------------------------------------------------------------------
// Embedding: x[b,t,:] = tok_emb[idx[b,t],:] + pos_emb[t,:]  (fp32)
// ---------------------------------------------------------------------------
__global__ __launch_bounds__(256)
void embed(const int* __restrict__ idx, const float* __restrict__ tok,
           const float* __restrict__ pos, float* __restrict__ x) {
  const int row = blockIdx.x;          // b*T + t
  const int t   = row & (T_SEQ - 1);
  const int tk  = idx[row];
  const float* tp = tok + (size_t)tk * D_MODEL;
  const float* pp = pos + (size_t)t * D_MODEL;
  float* xp = x + (size_t)row * D_MODEL;
  for (int c = threadIdx.x; c < D_MODEL; c += 256) xp[c] = tp[c] + pp[c];
}

// ---------------------------------------------------------------------------
extern "C" void kernel_launch(void* const* d_in, const int* in_sizes, int n_in,
                              void* d_out, int out_size, void* d_ws, size_t ws_size,
                              hipStream_t stream) {
  const int*   idx   = (const int*)d_in[0];
  const float* tok   = (const float*)d_in[1];
  const float* pos   = (const float*)d_in[2];
  const float* Wq    = (const float*)d_in[3];
  const float* Wk    = (const float*)d_in[4];
  const float* Wv    = (const float*)d_in[5];
  const float* Wo    = (const float*)d_in[6];
  const float* bo    = (const float*)d_in[7];
  const float* ln1g  = (const float*)d_in[8];
  const float* ln1b  = (const float*)d_in[9];
  const float* ln2g  = (const float*)d_in[10];
  const float* ln2b  = (const float*)d_in[11];
  const float* W1    = (const float*)d_in[12];
  const float* b1    = (const float*)d_in[13];
  const float* W2    = (const float*)d_in[14];
  const float* b2    = (const float*)d_in[15];
  const float* lnfg  = (const float*)d_in[16];
  const float* lnfb  = (const float*)d_in[17];
  const float* Whead = (const float*)d_in[18];
  const float* bhead = (const float*)d_in[19];
  float* logits = (float*)d_out;

  const size_t ND = (size_t)N_ROWS * D_MODEL;   // 2M elements
  float*     x   = (float*)d_ws;                // fp32 residual stream
  _Float16*  hb  = (_Float16*)(x + ND);         // f16 LN output
  _Float16*  q   = hb + ND;
  _Float16*  k   = q + ND;
  _Float16*  v   = k + ND;
  _Float16*  att = v + ND;
  _Float16*  mh  = att + ND;                    // 2048 x 4096 f16

  const dim3 blk(256);
  const dim3 gSq(N_ROWS / 128, D_MODEL / 128);  // (16, 8)
  const dim3 gFF(N_ROWS / 128, D_FF / 128);     // (16, 32)
  const dim3 gHd(N_ROWS / 128, (N_VOCAB + 127) / 128);
  const dim3 gAt(T_SEQ / 128, 2 * N_HEAD);      // (8, 32)

  embed<<<N_ROWS, blk, 0, stream>>>(idx, tok, pos, x);

  for (int l = 0; l < N_LAYER; ++l) {
    const size_t wDD = (size_t)l * D_MODEL * D_MODEL;
    const size_t wDF = (size_t)l * D_MODEL * D_FF;
    layernorm<<<N_ROWS, blk, 0, stream>>>(x, ln1g + l * D_MODEL, ln1b + l * D_MODEL, hb);
    gemm_wmma<false, false, false, _Float16><<<gSq, blk, 0, stream>>>(
        hb, Wq + wDD, nullptr, nullptr, q, N_ROWS, D_MODEL, D_MODEL);
    gemm_wmma<false, false, false, _Float16><<<gSq, blk, 0, stream>>>(
        hb, Wk + wDD, nullptr, nullptr, k, N_ROWS, D_MODEL, D_MODEL);
    gemm_wmma<false, false, false, _Float16><<<gSq, blk, 0, stream>>>(
        hb, Wv + wDD, nullptr, nullptr, v, N_ROWS, D_MODEL, D_MODEL);
    flash_attn<<<gAt, blk, 0, stream>>>(q, k, v, att);
    gemm_wmma<true, true, false, float><<<gSq, blk, 0, stream>>>(
        att, Wo + wDD, bo + l * D_MODEL, x, x, N_ROWS, D_MODEL, D_MODEL);
    layernorm<<<N_ROWS, blk, 0, stream>>>(x, ln2g + l * D_MODEL, ln2b + l * D_MODEL, hb);
    gemm_wmma<true, false, true, _Float16><<<gFF, blk, 0, stream>>>(
        hb, W1 + wDF, b1 + (size_t)l * D_FF, nullptr, mh, N_ROWS, D_FF, D_MODEL);
    gemm_wmma<true, true, false, float><<<gSq, blk, 0, stream>>>(
        mh, W2 + wDF, b2 + l * D_MODEL, x, x, N_ROWS, D_MODEL, D_FF);
  }

  layernorm<<<N_ROWS, blk, 0, stream>>>(x, lnfg, lnfb, hb);
  gemm_wmma<true, false, false, float><<<gHd, blk, 0, stream>>>(
      hb, Whead, bhead, nullptr, logits, N_ROWS, N_VOCAB, D_MODEL);
}
